// GraphIpaFrameDenoisingLayer_32152125178216
// MI455X (gfx1250) — matrix operand order repaired and checked by reading, not compiled, not run on hardware
//
#include <hip/hip_runtime.h>
#include <math.h>

#define NN   4096
#define EE   81920
#define EES  163840
#define HH   12

typedef __attribute__((ext_vector_type(16))) __bf16          v16bf;
typedef __attribute__((ext_vector_type(16))) unsigned short  v16u;
typedef __attribute__((ext_vector_type(8)))  float           v8f;
typedef __attribute__((ext_vector_type(4)))  unsigned int    u32x4;
typedef __attribute__((ext_vector_type(8)))  int             i32x8;
typedef __attribute__((ext_vector_type(4)))  int             i32x4;

#if defined(__gfx1250__) && __has_builtin(__builtin_amdgcn_tensor_load_to_lds) && __has_builtin(__builtin_amdgcn_s_wait_tensorcnt)
#define USE_TDM 1
#else
#define USE_TDM 0
#endif

struct Seg { const float* p; const int* idx; int w; };

static inline Seg mkseg(const float* p, int w, const int* idx = nullptr) {
  Seg s; s.p = p; s.idx = idx; s.w = w; return s;
}
static inline Seg noseg() { Seg s; s.p = nullptr; s.idx = nullptr; s.w = 0; return s; }

__device__ __forceinline__ unsigned short f2bf_bits(float f) {
  unsigned u = __builtin_bit_cast(unsigned, f);
  unsigned r = u + 0x7FFFu + ((u >> 16) & 1u);
  return (unsigned short)(r >> 16);
}

__device__ __forceinline__ unsigned long long pack4bf(float a, float b, float c, float d) {
  return (unsigned long long)f2bf_bits(a)
       | ((unsigned long long)f2bf_bits(b) << 16)
       | ((unsigned long long)f2bf_bits(c) << 32)
       | ((unsigned long long)f2bf_bits(d) << 48);
}

// ---------------------------------------------------------------------------
// Weight pre-pack: f32 W[K x Nc] -> bf16 fragment-ready 32x64 tiles.
// Block b = kt * ntn + nt holds 2048 bf16 laid out [ct(4)][lane(32)][e(16)]:
//   k = kt*32 + e + 16*(lane>>4),  n = nt*64 + ct*16 + (lane&15)
// ---------------------------------------------------------------------------
__global__ void pack_w(const float* __restrict__ W, unsigned short* __restrict__ Wp,
                       int K, int Nc) {
  int ntn = (Nc + 63) >> 6;
  long total = (long)((K + 31) >> 5) * ntn * 2048;
  long idx = (long)blockIdx.x * 256 + threadIdx.x;
  if (idx >= total) return;
  int  e    = (int)(idx & 15);
  int  lane = (int)((idx >> 4) & 31);
  int  ct   = (int)((idx >> 9) & 3);
  long blkid = idx >> 11;
  int  nt = (int)(blkid % ntn);
  int  kt = (int)(blkid / ntn);
  int  k  = kt * 32 + e + 16 * (lane >> 4);
  int  n  = nt * 64 + ct * 16 + (lane & 15);
  float v = (k < K && n < Nc) ? W[(long)k * Nc + n] : 0.f;
  Wp[idx] = f2bf_bits(v);
}

// ---------------------------------------------------------------------------
// B-tile stage: copy one contiguous packed 4KB block into LDS.
// TDM path: wave 0 issues a 1-D tensor_load_to_lds (512 x 8B elements),
// waits TENSORcnt, then the workgroup barrier publishes the tile.
// ---------------------------------------------------------------------------
__device__ __forceinline__ void stage_B(unsigned short* BsmF,
                                        const unsigned short* __restrict__ src,
                                        int tid) {
#if USE_TDM
  if (tid < 32) {   // wave 0 only (wave-uniform branch); TDM ignores EXEC
    unsigned ldsoff = (unsigned)(unsigned long long)
        (__attribute__((address_space(3))) const void*)BsmF;
    unsigned long long ga = (unsigned long long)src;
    u32x4 g0;
    g0[0] = 1u;                                   // count=1, user descriptor
    g0[1] = ldsoff;                               // lds_addr
    g0[2] = (unsigned)ga;                         // global_addr lo
    g0[3] = (unsigned)((ga >> 32) & 0x01FFFFFFu)  // global_addr hi
          | 0x80000000u;                          // type=2 ("image")
    i32x8 g1;
    g1[0] = 0x00030000;          // workgroup_mask=0, data_size=3 (8B)
    g1[1] = (int)0x02000000;     // tensor_dim0 lo16 (=512) << 16
    g1[2] = 0x00010000;          // tensor_dim0 hi16=0 | tensor_dim1(=1) lo16 << 16
    g1[3] = (int)0x02000000;     // tensor_dim1 hi16=0 | tile_dim0(=512) << 16
    g1[4] = 0;                   // tile_dim1=0 (unused), tile_dim2=0
    g1[5] = 512;                 // tensor_dim0_stride lo32
    g1[6] = 0;                   // tensor_dim0_stride hi16, dim1_stride lo16
    g1[7] = 0;
    i32x4 z4; z4[0] = 0; z4[1] = 0; z4[2] = 0; z4[3] = 0;
    i32x8 z8; z8[0] = 0; z8[1] = 0; z8[2] = 0; z8[3] = 0;
    z8[4] = 0; z8[5] = 0; z8[6] = 0; z8[7] = 0;
    __builtin_amdgcn_tensor_load_to_lds(g0, g1, z4, z4, z8, 0);
    __builtin_amdgcn_s_wait_tensorcnt((short)0);
  }
#else
  *(uint4*)(BsmF + tid * 8) = *(const uint4*)(src + tid * 8);
#endif
}

// ---------------------------------------------------------------------------
// WMMA GEMM: block tile 128x64, 8 waves; each wave owns one 16-row tile and
// computes against all four 16-col B tiles -> 4 WMMAs per wave per K-step.
// K and all segment widths must be multiples of 4 (true for this model), so
// A staging is pure float4-quad-or-zero: no scalar tails, no exec cascades.
// ---------------------------------------------------------------------------
#define BM 128
#define BN 64
#define KC 32

__global__ __launch_bounds__(256) void gemm_wmma_dense(
    const float* __restrict__ A, const unsigned short* __restrict__ Wp,
    const float* __restrict__ bias, float* __restrict__ C,
    int M, int K, int Nc, int act)
{
  __shared__ __align__(32) unsigned short AsmF[8][32][16];
  __shared__ __align__(32) unsigned short BsmF[2048];

  const int tid  = threadIdx.x;
  const int wm   = tid >> 5;          // wave id == row tile 0..7
  const int lane = tid & 31;
  const int half = lane >> 4;
  const int lm   = lane & 15;
  const int rowBase = blockIdx.y * BM;
  const int colBase = blockIdx.x * BN;
  const int ntn  = (Nc + 63) >> 6;

  v8f acc[4] = {};

  for (int k0 = 0; k0 < K; k0 += KC) {
    // A tile: 128 rows x 8 quads; one float4 load + one b64 LDS store each
#pragma unroll
    for (int i = 0; i < 4; ++i) {
      int qidx = tid + i * 256;        // 0..1023
      int r  = qidx >> 3;              // 0..127
      int c  = (qidx & 7) * 4;         // 0,4,..,28
      int gr = rowBase + r;
      int gk = k0 + c;
      float4 v = make_float4(0.f, 0.f, 0.f, 0.f);
      if (gr < M && gk < K)            // K % 4 == 0: quad fully in-bounds
        v = *(const float4*)(A + (long)gr * K + gk);
      int wt = r >> 4, rl = r & 15;
      int hf = c >> 4;
      int e0 = c & 15;
      *(unsigned long long*)&AsmF[wt][hf * 16 + rl][e0] = pack4bf(v.x, v.y, v.z, v.w);
    }
    // B tile: contiguous packed 4KB block
    stage_B(BsmF, Wp + (((long)(k0 >> 5) * ntn + (colBase >> 6)) << 11), tid);
    __syncthreads();

    v16u au = *(const v16u*)&AsmF[wm][lane][0];
    v16bf a = __builtin_bit_cast(v16bf, au);
#pragma unroll
    for (int ct = 0; ct < 4; ++ct) {
      v16u bu = *(const v16u*)&BsmF[(ct * 32 + lane) * 16];
      v16bf b = __builtin_bit_cast(v16bf, bu);
      acc[ct] = __builtin_amdgcn_wmma_f32_16x16x32_bf16(false, a, false, b, (short)0, acc[ct], false, false);
    }
    __syncthreads();
  }

#pragma unroll
  for (int ct = 0; ct < 4; ++ct) {
    int col = colBase + ct * 16 + lm;
    if (col >= Nc) continue;
    float bv = bias ? bias[col] : 0.f;
#pragma unroll
    for (int v = 0; v < 8; ++v) {
      int row = rowBase + wm * 16 + v + 8 * half;
      if (row < M) {
        float o = acc[ct][v] + bv;
        if (act == 1) o = fmaxf(o, 0.f);
        C[(long)row * Nc + col] = o;
      }
    }
  }
}

// ---------------------------------------------------------------------------
// WMMA GEMM, gathered/concatenated A (up to 3 segments, per-row index).
// All segment widths are multiples of 4, so each quad lies in one segment.
// ---------------------------------------------------------------------------
__global__ __launch_bounds__(256) void gemm_wmma_gather(
    Seg s0, Seg s1, Seg s2,
    const unsigned short* __restrict__ Wp, const float* __restrict__ bias,
    float* __restrict__ C, int M, int K, int Nc, int act)
{
  __shared__ __align__(32) unsigned short AsmF[8][32][16];
  __shared__ __align__(32) unsigned short BsmF[2048];

  const int tid  = threadIdx.x;
  const int wm   = tid >> 5;
  const int lane = tid & 31;
  const int half = lane >> 4;
  const int lm   = lane & 15;
  const int rowBase = blockIdx.y * BM;
  const int colBase = blockIdx.x * BN;
  const int ntn  = (Nc + 63) >> 6;
  const int w01 = s0.w + s1.w;

  v8f acc[4] = {};

  for (int k0 = 0; k0 < K; k0 += KC) {
#pragma unroll
    for (int i = 0; i < 4; ++i) {
      int qidx = tid + i * 256;
      int r  = qidx >> 3;
      int c  = (qidx & 7) * 4;
      int gr = rowBase + r;
      int gk = k0 + c;
      float4 v = make_float4(0.f, 0.f, 0.f, 0.f);
      if (gr < M && gk < K) {
        const float* p; const int* ix; int col; int wseg;
        if (gk < s0.w)      { p = s0.p; ix = s0.idx; col = gk;        wseg = s0.w; }
        else if (gk < w01)  { p = s1.p; ix = s1.idx; col = gk - s0.w; wseg = s1.w; }
        else                { p = s2.p; ix = s2.idx; col = gk - w01;  wseg = s2.w; }
        long rr = ix ? (long)ix[gr] : (long)gr;
        v = *(const float4*)(p + rr * (long)wseg + col);
      }
      int wt = r >> 4, rl = r & 15;
      int hf = c >> 4;
      int e0 = c & 15;
      *(unsigned long long*)&AsmF[wt][hf * 16 + rl][e0] = pack4bf(v.x, v.y, v.z, v.w);
    }
    stage_B(BsmF, Wp + (((long)(k0 >> 5) * ntn + (colBase >> 6)) << 11), tid);
    __syncthreads();

    v16u au = *(const v16u*)&AsmF[wm][lane][0];
    v16bf a = __builtin_bit_cast(v16bf, au);
#pragma unroll
    for (int ct = 0; ct < 4; ++ct) {
      v16u bu = *(const v16u*)&BsmF[(ct * 32 + lane) * 16];
      v16bf b = __builtin_bit_cast(v16bf, bu);
      acc[ct] = __builtin_amdgcn_wmma_f32_16x16x32_bf16(false, a, false, b, (short)0, acc[ct], false, false);
    }
    __syncthreads();
  }

#pragma unroll
  for (int ct = 0; ct < 4; ++ct) {
    int col = colBase + ct * 16 + lm;
    if (col >= Nc) continue;
    float bv = bias ? bias[col] : 0.f;
#pragma unroll
    for (int v = 0; v < 8; ++v) {
      int row = rowBase + wm * 16 + v + 8 * half;
      if (row < M) {
        float o = acc[ct][v] + bv;
        if (act == 1) o = fmaxf(o, 0.f);
        C[(long)row * Nc + col] = o;
      }
    }
  }
}

// ---------------------------------------------------------------------------
// LayerNorm:  out = ( LN(base + add*rowscale) * g + b ) * post
// ---------------------------------------------------------------------------
__global__ __launch_bounds__(128) void lnorm_kernel(
    const float* __restrict__ base, const float* __restrict__ add,
    const float* __restrict__ rowscale, int rs_mode,
    const float* __restrict__ post,
    const float* __restrict__ g, const float* __restrict__ b,
    float* __restrict__ out, int M, int Cc)
{
  __shared__ float red[128];
  const int row = blockIdx.x;
  const int tid = threadIdx.x;
  float rs = 1.f;
  if (rowscale) { float m = rowscale[row]; rs = (rs_mode == 2) ? (1.f - m) : m; }
  const long rb = (long)row * Cc;

  float s = 0.f;
  for (int c = tid; c < Cc; c += blockDim.x) {
    float x = base[rb + c];
    if (add) x += add[rb + c] * rs;
    s += x;
  }
  red[tid] = s; __syncthreads();
  for (int o = blockDim.x >> 1; o > 0; o >>= 1) { if (tid < o) red[tid] += red[tid + o]; __syncthreads(); }
  const float mean = red[0] / Cc;
  __syncthreads();

  float s2 = 0.f;
  for (int c = tid; c < Cc; c += blockDim.x) {
    float x = base[rb + c];
    if (add) x += add[rb + c] * rs;
    float d = x - mean; s2 += d * d;
  }
  red[tid] = s2; __syncthreads();
  for (int o = blockDim.x >> 1; o > 0; o >>= 1) { if (tid < o) red[tid] += red[tid + o]; __syncthreads(); }
  const float inv = rsqrtf(red[0] / Cc + 1e-5f);
  const float pm = post ? post[row] : 1.f;

  for (int c = tid; c < Cc; c += blockDim.x) {
    float x = base[rb + c];
    if (add) x += add[rb + c] * rs;
    out[rb + c] = ((x - mean) * inv * g[c] + b[c]) * pm;
  }
}

static void lnorm_launch(hipStream_t st, const float* base, const float* add,
                         const float* rowscale, int rs_mode, const float* post,
                         const float* g, const float* b, float* out, int M, int C) {
  lnorm_kernel<<<M, 128, 0, st>>>(base, add, rowscale, rs_mode, post, g, b, out, M, C);
}

// ---------------------------------------------------------------------------
__global__ void mask2f(const unsigned char* __restrict__ m, float* __restrict__ o, int n, int inv) {
  int i = blockIdx.x * blockDim.x + threadIdx.x;
  if (i < n) { float v = m[i] ? 1.f : 0.f; o[i] = inv ? 1.f - v : v; }
}

__global__ void rowscale_kernel(const float* __restrict__ in, const float* __restrict__ rs,
                                float* __restrict__ out, long rows, int C) {
  long idx = (long)blockIdx.x * blockDim.x + threadIdx.x;
  if (idx < rows * C) out[idx] = in[idx] * rs[idx / C];
}

// pts: (N, HP, 3) -> out = R*p + t
__global__ void glob_pts(const float* __restrict__ pts, const float* __restrict__ R,
                         const float* __restrict__ t, float* __restrict__ out,
                         int n_nodes, int HP) {
  int i = blockIdx.x * blockDim.x + threadIdx.x;
  if (i >= n_nodes * HP) return;
  int n = i / HP;
  const float* p = pts + (long)i * 3;
  const float* r = R + (long)n * 9;
  float x = p[0], y = p[1], z = p[2];
  out[(long)i * 3 + 0] = r[0]*x + r[1]*y + r[2]*z + t[n*3+0];
  out[(long)i * 3 + 1] = r[3]*x + r[4]*y + r[5]*z + t[n*3+1];
  out[(long)i * 3 + 2] = r[6]*x + r[7]*y + r[8]*z + t[n*3+2];
}

// attention logits per (edge, head)
__global__ void att_kernel(const float* __restrict__ q, const float* __restrict__ k,
                           const float* __restrict__ qpg, const float* __restrict__ kpg,
                           const float* __restrict__ biasz, const float* __restrict__ hw_raw,
                           const float* __restrict__ maskf,
                           const int* __restrict__ ei, const int* __restrict__ ej,
                           float* __restrict__ att, int Ecnt) {
  int idx = blockIdx.x * blockDim.x + threadIdx.x;
  if (idx >= Ecnt * HH) return;
  int e = idx / HH, h = idx % HH;
  int i = ei[e], j = ej[e];
  const float* qi = q + (long)i * 192 + h * 16;
  const float* kj = k + (long)j * 192 + h * 16;
  float dot = 0.f;
#pragma unroll
  for (int c = 0; c < 16; ++c) dot += qi[c] * kj[c];
  float a = dot * 0.14433756729740643f;
  a += 0.5773502691896258f * biasz[(long)e * HH + h];
  const float* qp = qpg + (long)i * 144 + h * 12;
  const float* kp = kpg + (long)j * 144 + h * 12;
  float d2 = 0.f;
#pragma unroll
  for (int c = 0; c < 12; ++c) { float d = qp[c] - kp[c]; d2 += d * d; }
  float w  = hw_raw[h];
  float sp = (w > 20.f) ? w : log1pf(expf(w));
  a -= 0.5f * sp * 0.13608276348795434f * d2;
  a += 100000.0f * (maskf[j] - 1.f);
  att[(long)e * HH + h] = a;
}

// contiguous fixed-degree segment softmax: one wave32 per (node, head)
__global__ __launch_bounds__(384) void seg_softmax(float* __restrict__ att, int deg) {
  int n    = blockIdx.x;
  int h    = threadIdx.x >> 5;
  int lane = threadIdx.x & 31;
  long base = (long)n * deg;
  float v0 = -1e30f, v1 = -1e30f;
  if (lane < deg)      v0 = att[(base + lane) * HH + h];
  if (lane + 32 < deg) v1 = att[(base + lane + 32) * HH + h];
  float m = fmaxf(v0, v1);
  for (int o = 16; o > 0; o >>= 1) m = fmaxf(m, __shfl_xor(m, o, 32));
  float x0 = (lane < deg)      ? __expf(v0 - m) : 0.f;
  float x1 = (lane + 32 < deg) ? __expf(v1 - m) : 0.f;
  float s = x0 + x1;
  for (int o = 16; o > 0; o >>= 1) s += __shfl_xor(s, o, 32);
  float inv = 1.f / (s + 1e-9f);
  if (lane < deg)      att[(base + lane) * HH + h]      = x0 * inv;
  if (lane + 32 < deg) att[(base + lane + 32) * HH + h] = x1 * inv;
}

__global__ void seg_accum_v(const float* __restrict__ a, const float* __restrict__ val,
                            const int* __restrict__ ej, float* __restrict__ out,
                            int n_nodes, int deg, int C, int out_stride, int out_off) {
  long idx = (long)blockIdx.x * blockDim.x + threadIdx.x;
  long total = (long)n_nodes * HH * C;
  if (idx >= total) return;
  int c = idx % C;
  int h = (int)((idx / C) % HH);
  int n = (int)(idx / ((long)C * HH));
  long eb = (long)n * deg;
  float s = 0.f;
  for (int d = 0; d < deg; ++d) {
    long e = eb + d;
    int j = ej[e];
    s += a[e * HH + h] * val[(long)j * HH * C + h * C + c];
  }
  out[(long)n * out_stride + out_off + h * C + c] = s;
}

__global__ void seg_accum_z(const float* __restrict__ a, const float* __restrict__ z,
                            float* __restrict__ out, int n_nodes, int deg, int CZ,
                            int out_stride, int out_off) {
  long idx = (long)blockIdx.x * blockDim.x + threadIdx.x;
  long total = (long)n_nodes * HH * CZ;
  if (idx >= total) return;
  int c = idx % CZ;
  int h = (int)((idx / CZ) % HH);
  int n = (int)(idx / ((long)CZ * HH));
  long eb = (long)n * deg;
  float s = 0.f;
  for (int d = 0; d < deg; ++d) {
    long e = eb + d;
    s += a[e * HH + h] * z[e * CZ + c];
  }
  out[(long)n * out_stride + out_off + h * CZ + c] = s;
}

__global__ void opt_finalize(const float* __restrict__ optraw, const float* __restrict__ R,
                             const float* __restrict__ t, float* __restrict__ concat,
                             int n_nodes) {
  int idx = blockIdx.x * blockDim.x + threadIdx.x;
  if (idx >= n_nodes * 96) return;
  int n = idx / 96, hp = idx % 96;
  const float* p = optraw + (long)n * 288 + hp * 3;
  const float* r = R + (long)n * 9;
  float x = p[0] - t[n*3+0], y = p[1] - t[n*3+1], z = p[2] - t[n*3+2];
  float ox = r[0]*x + r[3]*y + r[6]*z;
  float oy = r[1]*x + r[4]*y + r[7]*z;
  float oz = r[2]*x + r[5]*y + r[8]*z;
  float* dst = concat + (long)n * 2112 + 192 + hp * 3;
  dst[0] = ox; dst[1] = oy; dst[2] = oz;
  concat[(long)n * 2112 + 480 + hp] = sqrtf(ox*ox + oy*oy + oz*oz + 1e-8f);
}

__global__ void frame_kernel(const float* __restrict__ u6, const float* __restrict__ rn,
                             const float* __restrict__ R, const float* __restrict__ t,
                             float* __restrict__ rots_out, float* __restrict__ trans_out,
                             int n_nodes) {
  int n = blockIdx.x * blockDim.x + threadIdx.x;
  if (n >= n_nodes) return;
  float m = rn[n];
  float u0 = u6[n*6+0]*m, u1 = u6[n*6+1]*m, u2 = u6[n*6+2]*m;
  float u3 = u6[n*6+3]*m, u4 = u6[n*6+4]*m, u5 = u6[n*6+5]*m;
  float qw = 1.f, qx = u0, qy = u1, qz = u2;
  float inv = rsqrtf(qw*qw + qx*qx + qy*qy + qz*qz);
  qw *= inv; qx *= inv; qy *= inv; qz *= inv;
  float ru[9];
  ru[0] = 1 - 2*(qy*qy + qz*qz); ru[1] = 2*(qx*qy - qw*qz); ru[2] = 2*(qx*qz + qw*qy);
  ru[3] = 2*(qx*qy + qw*qz); ru[4] = 1 - 2*(qx*qx + qz*qz); ru[5] = 2*(qy*qz - qw*qx);
  ru[6] = 2*(qx*qz - qw*qy); ru[7] = 2*(qy*qz + qw*qx); ru[8] = 1 - 2*(qx*qx + qy*qy);
  const float* r = R + (long)n * 9;
  float* ro = rots_out + (long)n * 9;
#pragma unroll
  for (int a = 0; a < 3; ++a)
#pragma unroll
    for (int b = 0; b < 3; ++b) {
      float s = 0.f;
#pragma unroll
      for (int c = 0; c < 3; ++c) s += r[a*3+c] * ru[c*3+b];
      ro[a*3+b] = s;
    }
  trans_out[n*3+0] = r[0]*u3 + r[1]*u4 + r[2]*u5 + t[n*3+0];
  trans_out[n*3+1] = r[3]*u3 + r[4]*u4 + r[5]*u5 + t[n*3+1];
  trans_out[n*3+2] = r[6]*u3 + r[7]*u4 + r[8]*u5 + t[n*3+2];
}

__global__ void chi_kernel(const float* __restrict__ angraw, const float* __restrict__ chis_in,
                           const float* __restrict__ cm, const float* __restrict__ sm,
                           float* __restrict__ out, int n_nodes) {
  int idx = blockIdx.x * blockDim.x + threadIdx.x;
  if (idx >= n_nodes * 80) return;
  int n = idx / 80, q = idx % 80;
  float ang = tanhf(angraw[idx]) * 3.14159265358979323846f;
  float c = __cosf(ang), s = __sinf(ang);
  float x0 = chis_in[(long)n * 160 + q * 2 + 0];
  float x1 = chis_in[(long)n * 160 + q * 2 + 1];
  float n0 = c * x0 + s * x1;
  float n1 = -s * x0 + c * x1;
  float cmv = cm[n], smv = sm[n];
  out[(long)n * 160 + q * 2 + 0] = (cmv != 0.f ? n0 : x0) * smv;
  out[(long)n * 160 + q * 2 + 1] = (cmv != 0.f ? n1 : x1) * smv;
}

__global__ __launch_bounds__(32) void seqtrans_kernel(
    const float* __restrict__ tmraw, const float* __restrict__ probs,
    const float* __restrict__ snm, const float* __restrict__ sm,
    float* __restrict__ sp_out, int n_nodes) {
  __shared__ float tm[20][21];
  int n = blockIdx.x;
  int t = threadIdx.x;
  if (t < 20) {
    float row[20];
    float mx = -1e30f;
    for (int j = 0; j < 20; ++j) {
      float v = tmraw[(long)n * 400 + t * 20 + j] + (t == j ? 1.f : 0.f);
      row[j] = v; mx = fmaxf(mx, v);
    }
    float s = 0.f;
    for (int j = 0; j < 20; ++j) { row[j] = __expf(row[j] - mx); s += row[j]; }
    float inv = 1.f / s;
    bool noise = snm[n] != 0.f;
    for (int j = 0; j < 20; ++j)
      tm[t][j] = noise ? row[j] * inv : (t == j ? 1.f : 0.f);
  }
  __syncthreads();
  if (t < 20) {
    float s = 0.f;
    for (int i2 = 0; i2 < 20; ++i2) s += probs[(long)n * 20 + i2] * tm[i2][t];
    sp_out[(long)n * 20 + t] = s * sm[n];
  }
}

__global__ void add_gather3(float* __restrict__ dst, Seg s0, Seg s1, Seg s2,
                            long rows, int K) {
  long idx = (long)blockIdx.x * blockDim.x + threadIdx.x;
  if (idx >= rows * K) return;
  long r = idx / K; int k = (int)(idx % K);
  const float* p; const int* ix; int col, wseg;
  if (k < s0.w)              { p = s0.p; ix = s0.idx; col = k;               wseg = s0.w; }
  else if (k < s0.w + s1.w)  { p = s1.p; ix = s1.idx; col = k - s0.w;        wseg = s1.w; }
  else                       { p = s2.p; ix = s2.idx; col = k - s0.w - s1.w; wseg = s2.w; }
  long rr = ix ? (long)ix[r] : r;
  dst[idx] += p[rr * (long)wseg + col];
}

// ---------------------------------------------------------------------------
extern "C" void kernel_launch(void* const* d_in, const int* in_sizes, int n_in,
                              void* d_out, int out_size, void* d_ws, size_t ws_size,
                              hipStream_t stream) {
  (void)in_sizes; (void)n_in; (void)out_size; (void)ws_size;
  hipStream_t st = stream;
  auto F = [&](int i) { return (const float*)d_in[i]; };

  const float* node_features       = F(0);
  const float* rots                = F(1);
  const float* trans               = F(2);
  const float* seq_probs           = F(3);
  const float* chis                = F(4);
  const float* edge_features       = F(5);
  const int*   ei                  = (const int*)d_in[6];
  const int*   ej                  = ei + EE;
  const float* new_seq_edge_inputs = F(7);
  const float* seq_edge_features   = F(8);
  const int*   si                  = (const int*)d_in[9];
  const int*   sj                  = si + EES;
  const unsigned char* m_res = (const unsigned char*)d_in[10];
  const unsigned char* m_seq = (const unsigned char*)d_in[11];
  const unsigned char* m_rn  = (const unsigned char*)d_in[12];
  const unsigned char* m_sn  = (const unsigned char*)d_in[13];
  const unsigned char* m_cn  = (const unsigned char*)d_in[14];

  // ---- params (jax tree_flatten order: sorted dict keys; 'b'<'w', 'b'<'g') ----
  int pi = 15;
  auto nx = [&]() { return (const float*)d_in[pi++]; };
  const float *bb_b = nx(), *bb_w = nx();
  const float *chi_l1_b = nx(), *chi_l1_w = nx(), *chi_l2_b = nx(), *chi_l2_w = nx(),
              *chi_l3_b = nx(), *chi_l3_w = nx();
  const float *ee_l1_b = nx(), *ee_l1_w = nx(), *ee_l2_b = nx(), *ee_l2_w = nx(),
              *ee_l3_b = nx(), *ee_l3_w = nx(), *ee_ln_b = nx(), *ee_ln_g = nx();
  const float *initln_b = nx(), *initln_g = nx();
  struct IPA {
    const float *bias_b, *bias_w, *hw, *k_b, *k_w, *kp_b, *kp_w,
                *out_b, *out_w, *q_b, *q_w, *qp_b, *qp_w, *v_b, *v_w, *vp_b, *vp_w;
  };
  auto readIPA = [&]() {
    IPA p;
    p.bias_b = nx(); p.bias_w = nx(); p.hw = nx();
    p.k_b = nx(); p.k_w = nx(); p.kp_b = nx(); p.kp_w = nx();
    p.out_b = nx(); p.out_w = nx();
    p.q_b = nx(); p.q_w = nx(); p.qp_b = nx(); p.qp_w = nx();
    p.v_b = nx(); p.v_w = nx(); p.vp_b = nx(); p.vp_w = nx();
    return p;
  };
  IPA ipa_seq_p     = readIPA();   // 'ipa_seq' < 'ipa_spatial'
  IPA ipa_spatial_p = readIPA();
  const float *lns1_b = nx(), *lns1_g = nx();
  const float *lns2_b = nx(), *lns2_g = nx();
  const float *nt_l1_b = nx(), *nt_l1_w = nx(), *nt_l2_b = nx(), *nt_l2_w = nx(),
              *nt_l3_b = nx(), *nt_l3_w = nx(), *nt_ln_b = nx(), *nt_ln_g = nx();
  const float *scn_l1_b = nx(), *scn_l1_w = nx(), *scn_l2_b = nx(), *scn_l2_w = nx(),
              *scn_l3_b = nx(), *scn_l3_w = nx();
  const float *seln_b = nx(), *seln_g = nx();
  const float *set_fin_b = nx(), *set_fin_w = nx(), *set_init_b = nx(), *set_init_w = nx(),
              *set_ln_b = nx(), *set_ln_g = nx(), *set_t1_b = nx(), *set_t1_w = nx(),
              *set_t2_b = nx(), *set_t2_w = nx();
  const float *seu_l1_b = nx(), *seu_l1_w = nx(), *seu_l2_b = nx(), *seu_l2_w = nx(),
              *seu_l3_b = nx(), *seu_l3_w = nx();
  const float *st_l1_b = nx(), *st_l1_w = nx(), *st_l2_b = nx(), *st_l2_w = nx(),
              *st_l3_b = nx(), *st_l3_w = nx();

  // ---- output slices (tuple order) ----
  float* out      = (float*)d_out;
  float* out_nf   = out;
  float* out_rots = out_nf   + (long)NN * 384;
  float* out_tr   = out_rots + (long)NN * 9;
  float* out_sp   = out_tr   + (long)NN * 3;
  float* out_chis = out_sp   + (long)NN * 20;
  float* out_ef   = out_chis + (long)NN * 160;
  float* out_sef  = out_ef   + (long)EE * 128;

  // ---- workspace carve-up (32B-aligned bump allocator) ----
  float* ws = (float*)d_ws;
  size_t off = 0;
  auto alloc = [&](size_t nf) {
    off = (off + 7) & ~(size_t)7;
    float* p = ws + off; off += nf; return p;
  };
  float* fm_res     = alloc(NN);
  float* fm_res_inv = alloc(NN);
  float* fm_seq     = alloc(NN);
  float* fm_rn      = alloc(NN);
  float* fm_sn      = alloc(NN);
  float* fm_cn      = alloc(NN);
  float* nfbuf   = alloc((size_t)NN * 384);
  float* tA      = alloc((size_t)NN * 448);
  float* tB      = alloc((size_t)NN * 448);
  float* tC      = alloc((size_t)NN * 448);
  float* qbuf    = alloc((size_t)NN * 192);
  float* kbuf    = alloc((size_t)NN * 192);
  float* vbuf    = alloc((size_t)NN * 192);
  float* qpb     = alloc((size_t)NN * 144);
  float* kpb     = alloc((size_t)NN * 144);
  float* vpb     = alloc((size_t)NN * 288);
  float* qpg     = alloc((size_t)NN * 144);
  float* kpg     = alloc((size_t)NN * 144);
  float* vpg     = alloc((size_t)NN * 288);
  float* attb    = alloc((size_t)EES * 12);
  float* biasb   = alloc((size_t)EES * 12);
  float* optraw  = alloc((size_t)NN * 288);
  float* concatb = alloc((size_t)NN * 2112);
  float* updb    = alloc((size_t)NN * 384);
  float* hbuf    = alloc((size_t)NN * 192);
  float* sefbuf  = alloc((size_t)EES * 128);
  float* e1      = alloc((size_t)EES * 128);
  float* e2      = alloc((size_t)EES * 128);
  float* e3      = alloc((size_t)EES * 128);
  float* big1    = alloc((size_t)EES * 512);
  float* big2    = alloc((size_t)EES * 512);

  auto blk = [](long n) { return (unsigned)((n + 255) / 256); };

  // pack weights then launch the right GEMM variant
  auto gemmp = [&](Seg s0, Seg s1, Seg s2, const float* W, const float* b, float* Cc,
                   int M, int K, int Ncol, int act) {
    int kt = (K + 31) >> 5, nt = (Ncol + 63) >> 6;
    size_t cnt = (size_t)kt * nt * 2048;              // bf16 elements
    unsigned short* Wp = (unsigned short*)alloc(cnt / 2 + 8);
    pack_w<<<blk((long)cnt), 256, 0, st>>>(W, Wp, K, Ncol);
    dim3 g(nt, (M + BM - 1) / BM);
    if (s1.w == 0 && s0.idx == nullptr)
      gemm_wmma_dense<<<g, 256, 0, st>>>(s0.p, Wp, b, Cc, M, K, Ncol, act);
    else
      gemm_wmma_gather<<<g, 256, 0, st>>>(s0, s1, s2, Wp, b, Cc, M, K, Ncol, act);
  };

  // ---- masks -> float ----
  mask2f<<<blk(NN), 256, 0, st>>>(m_res, fm_res, NN, 0);
  mask2f<<<blk(NN), 256, 0, st>>>(m_res, fm_res_inv, NN, 1);
  mask2f<<<blk(NN), 256, 0, st>>>(m_seq, fm_seq, NN, 0);
  mask2f<<<blk(NN), 256, 0, st>>>(m_rn, fm_rn, NN, 0);
  mask2f<<<blk(NN), 256, 0, st>>>(m_sn, fm_sn, NN, 0);
  mask2f<<<blk(NN), 256, 0, st>>>(m_cn, fm_cn, NN, 0);

  // ---- stage 1: seq_chi_node ff3 + init layernorm ----
  gemmp(mkseg(node_features, 384), mkseg(chis, 160), mkseg(seq_probs, 20),
        scn_l1_w, scn_l1_b, tA, NN, 564, 384, 1);
  gemmp(mkseg(tA, 384), noseg(), noseg(), scn_l2_w, scn_l2_b, tB, NN, 384, 384, 1);
  gemmp(mkseg(tB, 384), noseg(), noseg(), scn_l3_w, scn_l3_b, tC, NN, 384, 384, 0);
  lnorm_launch(st, node_features, tC, fm_res, 2, nullptr, initln_g, initln_b, nfbuf, NN, 384);

  // ---- stage 2: edge_embed ----
  gemmp(mkseg(edge_features, 132), mkseg(nfbuf, 384, ei), mkseg(nfbuf, 384, ej),
        ee_l1_w, ee_l1_b, e1, EE, 900, 128, 1);
  gemmp(mkseg(e1, 128), noseg(), noseg(), ee_l2_w, ee_l2_b, e2, EE, 128, 128, 1);
  gemmp(mkseg(e2, 128), noseg(), noseg(), ee_l3_w, ee_l3_b, e3, EE, 128, 128, 0);
  lnorm_launch(st, e3, nullptr, nullptr, 0, nullptr, ee_ln_g, ee_ln_b, out_ef, EE, 128);

  // ---- stage 3: seq_edge_update ----
  gemmp(mkseg(new_seq_edge_inputs, 132), mkseg(nfbuf, 384, si), mkseg(nfbuf, 384, sj),
        seu_l1_w, seu_l1_b, e1, EES, 900, 128, 1);
  gemmp(mkseg(e1, 128), noseg(), noseg(), seu_l2_w, seu_l2_b, e2, EES, 128, 128, 1);
  gemmp(mkseg(e2, 128), noseg(), noseg(), seu_l3_w, seu_l3_b, e3, EES, 128, 128, 0);
  lnorm_launch(st, seq_edge_features, e3, nullptr, 0, nullptr, seln_g, seln_b, sefbuf, EES, 128);

  // ---- IPA (shared path) ----
  auto run_ipa = [&](const IPA& P, const float* z, const int* ai, const int* aj,
                     int Ecnt, int deg, const float* maskf) {
    gemmp(mkseg(nfbuf, 384), noseg(), noseg(), P.q_w,  P.q_b,  qbuf, NN, 384, 192, 0);
    gemmp(mkseg(nfbuf, 384), noseg(), noseg(), P.k_w,  P.k_b,  kbuf, NN, 384, 192, 0);
    gemmp(mkseg(nfbuf, 384), noseg(), noseg(), P.v_w,  P.v_b,  vbuf, NN, 384, 192, 0);
    gemmp(mkseg(nfbuf, 384), noseg(), noseg(), P.qp_w, P.qp_b, qpb,  NN, 384, 144, 0);
    gemmp(mkseg(nfbuf, 384), noseg(), noseg(), P.kp_w, P.kp_b, kpb,  NN, 384, 144, 0);
    gemmp(mkseg(nfbuf, 384), noseg(), noseg(), P.vp_w, P.vp_b, vpb,  NN, 384, 288, 0);
    glob_pts<<<blk((long)NN * 48), 256, 0, st>>>(qpb, rots, trans, qpg, NN, 48);
    glob_pts<<<blk((long)NN * 48), 256, 0, st>>>(kpb, rots, trans, kpg, NN, 48);
    glob_pts<<<blk((long)NN * 96), 256, 0, st>>>(vpb, rots, trans, vpg, NN, 96);
    gemmp(mkseg(z, 128), noseg(), noseg(), P.bias_w, P.bias_b, biasb, Ecnt, 128, 12, 0);
    att_kernel<<<blk((long)Ecnt * HH), 256, 0, st>>>(qbuf, kbuf, qpg, kpg, biasb, P.hw,
                                                     maskf, ai, aj, attb, Ecnt);
    seg_softmax<<<NN, 384, 0, st>>>(attb, deg);
    seg_accum_v<<<blk((long)NN * HH * 16), 256, 0, st>>>(attb, vbuf, aj, concatb, NN, deg, 16, 2112, 0);
    seg_accum_v<<<blk((long)NN * HH * 24), 256, 0, st>>>(attb, vpg, aj, optraw, NN, deg, 24, 288, 0);
    opt_finalize<<<blk((long)NN * 96), 256, 0, st>>>(optraw, rots, trans, concatb, NN);
    seg_accum_z<<<blk((long)NN * HH * 128), 256, 0, st>>>(attb, z, concatb, NN, deg, 128, 2112, 576);
    gemmp(mkseg(concatb, 2112), noseg(), noseg(), P.out_w, P.out_b, updb, NN, 2112, 384, 0);
  };

  // ---- stage 4: ipa_spatial (mask = ~res) + ln_s1 ----
  run_ipa(ipa_spatial_p, out_ef, ei, ej, EE, 20, fm_res_inv);
  lnorm_launch(st, nfbuf, updb, fm_res, 1, nullptr, lns1_g, lns1_b, nfbuf, NN, 384);

  // ---- stage 5: ipa_seq (mask = res) + ln_s2 ----
  run_ipa(ipa_seq_p, sefbuf, si, sj, EES, 40, fm_res);
  lnorm_launch(st, nfbuf, updb, fm_res, 1, nullptr, lns2_g, lns2_b, nfbuf, NN, 384);

  // ---- stage 6: node transition ff3 + ln (+ *rm) -> final nf in d_out ----
  gemmp(mkseg(nfbuf, 384), noseg(), noseg(), nt_l1_w, nt_l1_b, tA, NN, 384, 384, 1);
  gemmp(mkseg(tA, 384), noseg(), noseg(), nt_l2_w, nt_l2_b, tB, NN, 384, 384, 1);
  gemmp(mkseg(tB, 384), noseg(), noseg(), nt_l3_w, nt_l3_b, tC, NN, 384, 384, 0);
  lnorm_launch(st, nfbuf, tC, nullptr, 0, fm_res, nt_ln_g, nt_ln_b, out_nf, NN, 384);

  // ---- stage 7: backbone frame update ----
  rowscale_kernel<<<blk((long)NN * 384), 256, 0, st>>>(out_nf, fm_rn, tA, NN, 384);
  gemmp(mkseg(tA, 384), noseg(), noseg(), bb_w, bb_b, tB, NN, 384, 6, 0);
  frame_kernel<<<blk(NN), 256, 0, st>>>(tB, fm_rn, rots, trans, out_rots, out_tr, NN);

  // ---- stage 8: seq_edge_transition ----
  gemmp(mkseg(out_nf, 384), noseg(), noseg(), set_init_w, set_init_b, hbuf, NN, 384, 192, 0);
  gemmp(mkseg(sefbuf, 128), mkseg(hbuf, 192, si), mkseg(hbuf, 192, sj),
        set_t1_w, set_t1_b, big1, EES, 512, 512, 1);
  gemmp(mkseg(big1, 512), noseg(), noseg(), set_t2_w, set_t2_b, big2, EES, 512, 512, 1);
  add_gather3<<<blk((long)EES * 512), 256, 0, st>>>(big2, mkseg(sefbuf, 128),
                                                    mkseg(hbuf, 192, si), mkseg(hbuf, 192, sj),
                                                    EES, 512);
  gemmp(mkseg(big2, 512), noseg(), noseg(), set_fin_w, set_fin_b, e1, EES, 512, 128, 0);
  lnorm_launch(st, e1, nullptr, nullptr, 0, nullptr, set_ln_g, set_ln_b, out_sef, EES, 128);

  // ---- stage 9: chi angles ----
  gemmp(mkseg(out_nf, 384), noseg(), noseg(), chi_l1_w, chi_l1_b, tA, NN, 384, 80, 1);
  gemmp(mkseg(tA, 80), noseg(), noseg(), chi_l2_w, chi_l2_b, tB, NN, 80, 80, 1);
  gemmp(mkseg(tB, 80), noseg(), noseg(), chi_l3_w, chi_l3_b, tC, NN, 80, 80, 0);
  chi_kernel<<<blk((long)NN * 80), 256, 0, st>>>(tC, chis, fm_cn, fm_seq, out_chis, NN);

  // ---- stage 10: sequence transition ----
  gemmp(mkseg(out_nf, 384), noseg(), noseg(), st_l1_w, st_l1_b, tA, NN, 384, 384, 1);
  gemmp(mkseg(tA, 384), noseg(), noseg(), st_l2_w, st_l2_b, tB, NN, 384, 384, 1);
  gemmp(mkseg(tB, 384), noseg(), noseg(), st_l3_w, st_l3_b, tC, NN, 384, 400, 0);
  seqtrans_kernel<<<NN, 32, 0, st>>>(tC, seq_probs, fm_sn, fm_seq, out_sp, NN);
}
